// CapsuleLayer_70033736729252
// MI455X (gfx1250) — matrix operand order repaired and checked by reading, not compiled, not run on hardware
//
#include <hip/hip_runtime.h>

// ---------------------------------------------------------------------------
// CapsuleLayer dynamic routing, fused-GEMM formulation for gfx1250 (MI455X).
//   B=256 batch, IN=8, I=1152 in-caps, N=10 out-caps, O=16 cap dim.
//   s_j  = X(256x9216) * Wc_n(9216x16)   per n      -> v_wmma_f32_16x16x32_bf16
//   P    = X^T(9216x256) * V(256x160)               -> v_wmma_f32_16x16x32_bf16
//   u_vj = elementwise dot of W with P (tiny)
// u_hat (189 MB) is never materialized; working set ~19 MB stays L2-resident.
// GEMM1 uses an 8-way K-split (1280 waves) to fill the ~1024 SIMD32s.
// ---------------------------------------------------------------------------

typedef __attribute__((ext_vector_type(16))) __bf16 bf16x16;
typedef __attribute__((ext_vector_type(8)))  float  v8f;

#define B_    256
#define INSZ  8
#define I_    1152
#define N_    10
#define O_    16
#define IK    9216          // I_*INSZ
#define NO    160           // N_*O_
#define KS1   288           // IK/32  k-steps for GEMM1
#define MT1   16            // B_/16  m-tiles for GEMM1
#define MT2   576           // IK/16  m-tiles for GEMM2
#define KS2   8             // B_/32  k-steps for GEMM2
#define KSPLIT 8            // waves per GEMM1 tile (k-split factor)

// ISA 16-bit A-fragment (16x32) K index for (lane, element e of v16bf)
__device__ __forceinline__ int a_kt(int lane, int e) {
  int kt = (e < 8) ? e : e + 8;          // lanes 0-15: K 0-7 then 16-23
  return (lane < 16) ? kt : kt + 8;      // lanes 16-31: K 8-15 then 24-31
}
// ISA 16-bit B-fragment (32x16) K index: lanes 0-15 hold K 0-7,16-23; 16-31 hold 8-15,24-31
__device__ __forceinline__ int b_kt(int lane, int e) {
  int kt = (e < 8) ? e : e + 8;
  return (lane < 16) ? kt : kt + 8;
}

// ---- pack x (B,8,I) fp32 -> A-fragment stream for GEMM1: [mtile][ks][lane][16]
__global__ void pack_x_a1(const float* __restrict__ x, __bf16* __restrict__ XP) {
  int idx = blockIdx.x * blockDim.x + threadIdx.x;
  const int total = MT1 * KS1 * 32 * 16;
  if (idx >= total) return;
  int e = idx & 15, t = idx >> 4;
  int lane = t & 31; t >>= 5;
  int ks = t % KS1, mtile = t / KS1;
  int b  = mtile * 16 + (lane & 15);
  int kg = ks * 32 + a_kt(lane, e);
  int i  = kg >> 3, k = kg & 7;
  XP[idx] = (__bf16)x[(b * INSZ + k) * I_ + i];
}

// ---- pack x^T -> A-fragment stream for GEMM2: rows=(i,k), cols=b
__global__ void pack_x_a2(const float* __restrict__ x, __bf16* __restrict__ XTP) {
  int idx = blockIdx.x * blockDim.x + threadIdx.x;
  const int total = MT2 * KS2 * 32 * 16;
  if (idx >= total) return;
  int e = idx & 15, t = idx >> 4;
  int lane = t & 31; t >>= 5;
  int ks = t % KS2, mtile = t / KS2;
  int row = mtile * 16 + (lane & 15);        // ik
  int b   = ks * 32 + a_kt(lane, e);
  int i   = row >> 3, k = row & 7;
  XTP[idx] = (__bf16)x[(b * INSZ + k) * I_ + i];
}

// ---- build Wc = c[i,n]*0.03*W[i,n,o,k] in B-fragment stream: [n][ks][lane][16]
__global__ void pack_wc(const float* __restrict__ W, const float* __restrict__ c,
                        __bf16* __restrict__ WCP) {
  int idx = blockIdx.x * blockDim.x + threadIdx.x;
  const int total = N_ * KS1 * 32 * 16;
  if (idx >= total) return;
  int e = idx & 15, t = idx >> 4;
  int lane = t & 31; t >>= 5;
  int ks = t % KS1, n = t / KS1;
  int o  = lane & 15;
  int kg = ks * 32 + b_kt(lane, e);
  int i  = kg >> 3, k = kg & 7;
  float w = W[((i * N_ + n) * O_ + o) * INSZ + k];
  WCP[idx] = (__bf16)(c[i * N_ + n] * 0.03f * w);
}

// ---- pack v (B,N,O) fp32 -> B-fragment stream for GEMM2: [n][ks][lane][16]
__global__ void pack_v(const float* __restrict__ v, __bf16* __restrict__ VP) {
  int idx = blockIdx.x * blockDim.x + threadIdx.x;
  const int total = N_ * KS2 * 32 * 16;
  if (idx >= total) return;
  int e = idx & 15, t = idx >> 4;
  int lane = t & 31; t >>= 5;
  int ks = t % KS2, n = t / KS2;
  int o = lane & 15;
  int b = ks * 32 + b_kt(lane, e);
  VP[idx] = (__bf16)v[(b * N_ + n) * O_ + o];
}

// ---- GEMM1: s(256x160) = X(256x9216) * Wc(9216x160)
//      8-way K-split per 16x16 tile + LDS reduce; 160 blocks x 8 waves.
__global__ __launch_bounds__(32 * KSPLIT) void gemm_s(const __bf16* __restrict__ XP,
                                                      const __bf16* __restrict__ WCP,
                                                      float* __restrict__ s) {
  __shared__ float red[KSPLIT][32][8];
  const int mtile = blockIdx.x;            // 0..15
  const int n     = blockIdx.y;            // 0..9
  const int wave  = threadIdx.x >> 5;
  const int lane  = threadIdx.x & 31;
  const int kchunk = KS1 / KSPLIT;         // 36 k-steps per wave

  const bf16x16* Ap = (const bf16x16*)XP  + ((size_t)mtile * KS1 + wave * kchunk) * 32 + lane;
  const bf16x16* Bp = (const bf16x16*)WCP + ((size_t)n     * KS1 + wave * kchunk) * 32 + lane;

  v8f acc0 = {}, acc1 = {};
  #pragma unroll 4
  for (int ks = 0; ks < kchunk; ks += 2) {
    bf16x16 a0 = Ap[(size_t)ks * 32];
    bf16x16 b0 = Bp[(size_t)ks * 32];
    bf16x16 a1 = Ap[(size_t)(ks + 1) * 32];
    bf16x16 b1 = Bp[(size_t)(ks + 1) * 32];
    acc0 = __builtin_amdgcn_wmma_f32_16x16x32_bf16(false, a0, false, b0,
                                                   (short)0, acc0, false, false);
    acc1 = __builtin_amdgcn_wmma_f32_16x16x32_bf16(false, a1, false, b1,
                                                   (short)0, acc1, false, false);
  }
  v8f acc = acc0 + acc1;
  #pragma unroll
  for (int r = 0; r < 8; ++r) red[wave][lane][r] = acc[r];
  __syncthreads();
  if (wave == 0) {
    #pragma unroll
    for (int r = 0; r < 8; ++r) {
      float sum = 0.f;
      #pragma unroll
      for (int w = 0; w < KSPLIT; ++w) sum += red[w][lane][r];
      int row = mtile * 16 + ((lane < 16) ? r : 8 + r);
      int col = n * 16 + (lane & 15);
      s[row * NO + col] = sum;
    }
  }
}

// ---- GEMM2: P(9216x160) = X^T(9216x256) * V(256x160); one wave per 16x16 tile
__global__ __launch_bounds__(128) void gemm_p(const __bf16* __restrict__ XTP,
                                              const __bf16* __restrict__ VP,
                                              float* __restrict__ P) {
  const int wave = threadIdx.x >> 5;
  const int lane = threadIdx.x & 31;
  const int tile = blockIdx.x * 4 + wave;  // 0..5759
  const int mtile = tile % MT2;
  const int n     = tile / MT2;

  const bf16x16* Ap = (const bf16x16*)XTP + (size_t)mtile * KS2 * 32 + lane;
  const bf16x16* Bp = (const bf16x16*)VP  + (size_t)n     * KS2 * 32 + lane;

  v8f acc0 = {}, acc1 = {};
  #pragma unroll
  for (int ks = 0; ks < KS2; ks += 2) {
    bf16x16 a0 = Ap[(size_t)ks * 32];
    bf16x16 b0 = Bp[(size_t)ks * 32];
    bf16x16 a1 = Ap[(size_t)(ks + 1) * 32];
    bf16x16 b1 = Bp[(size_t)(ks + 1) * 32];
    acc0 = __builtin_amdgcn_wmma_f32_16x16x32_bf16(false, a0, false, b0,
                                                   (short)0, acc0, false, false);
    acc1 = __builtin_amdgcn_wmma_f32_16x16x32_bf16(false, a1, false, b1,
                                                   (short)0, acc1, false, false);
  }
  v8f acc = acc0 + acc1;
  #pragma unroll
  for (int r = 0; r < 8; ++r) {
    int row = mtile * 16 + ((lane < 16) ? r : 8 + r);
    int col = n * 16 + (lane & 15);
    P[row * NO + col] = acc[r];
  }
}

// ---- squash over O; optionally emit final output (B,N,O,1)
__global__ void squash_k(const float* __restrict__ s, float* __restrict__ v,
                         float* __restrict__ out) {
  int idx = blockIdx.x * blockDim.x + threadIdx.x;
  if (idx >= B_ * NO) return;
  int o = idx & 15, bn = idx >> 4;                 // bn = b*N_+n
  const float* sp = s + bn * 16;                   // s[b*160 + n*16 + o]
  float msq = 0.f;
  #pragma unroll
  for (int j = 0; j < 16; ++j) { float t = sp[j]; msq += t * t; }
  float mag = sqrtf(msq);
  float val = sp[o] * (msq / ((1.0f + msq) * mag));
  v[idx] = val;
  if (out) out[idx] = val;
}

// ---- agreement: b[i,n] += (0.03/B) * sum_{o,k} W[i,n,o,k] * P[(i,k),(n,o)]
__global__ void uvj_k(const float* __restrict__ W, const float* __restrict__ P,
                      float* __restrict__ bij) {
  int idx = blockIdx.x * blockDim.x + threadIdx.x;
  if (idx >= I_ * N_) return;
  int n = idx % N_, i = idx / N_;
  const float* wp = W + (size_t)(i * N_ + n) * O_ * INSZ;
  float sum = 0.f;
  for (int o = 0; o < O_; ++o) {
    const float* pp = P + (size_t)(i * INSZ) * NO + n * 16 + o;
    #pragma unroll
    for (int k = 0; k < INSZ; ++k)
      sum += wp[o * INSZ + k] * pp[(size_t)k * NO];
  }
  bij[idx] += sum * (0.03f / (float)B_);
}

// ---- softmax over N per input capsule i
__global__ void softmax_k(const float* __restrict__ bij, float* __restrict__ c) {
  int i = blockIdx.x * blockDim.x + threadIdx.x;
  if (i >= I_) return;
  float m = -3.4e38f;
  #pragma unroll
  for (int n = 0; n < N_; ++n) m = fmaxf(m, bij[i * N_ + n]);
  float e[N_], sum = 0.f;
  #pragma unroll
  for (int n = 0; n < N_; ++n) { e[n] = expf(bij[i * N_ + n] - m); sum += e[n]; }
  float inv = 1.0f / sum;
  #pragma unroll
  for (int n = 0; n < N_; ++n) c[i * N_ + n] = e[n] * inv;
}

__global__ void zero_k(float* __restrict__ p, int n) {
  int idx = blockIdx.x * blockDim.x + threadIdx.x;
  if (idx < n) p[idx] = 0.f;
}

extern "C" void kernel_launch(void* const* d_in, const int* in_sizes, int n_in,
                              void* d_out, int out_size, void* d_ws, size_t ws_size,
                              hipStream_t stream) {
  const float* x = (const float*)d_in[0];   // (256, 8, 1152)
  const float* W = (const float*)d_in[1];   // (1, 1152, 10, 16, 8)

  char* ws = (char*)d_ws;                   // ~18.8 MB carved below
  __bf16* XP  = (__bf16*)(ws + 0);          // 4,718,592 B  A-frag X
  __bf16* XTP = (__bf16*)(ws + 4718592);    // 4,718,592 B  A-frag X^T
  __bf16* WCP = (__bf16*)(ws + 9437184);    // 2,949,120 B  B-frag Wc
  __bf16* VP  = (__bf16*)(ws + 12386304);   //    81,920 B  B-frag V
  float*  s   = (float*)(ws + 12468224);    //   163,840 B
  float*  v   = (float*)(ws + 12632064);    //   163,840 B
  float*  P   = (float*)(ws + 12795904);    // 5,898,240 B
  float*  bij = (float*)(ws + 18694144);    //    46,080 B
  float*  c   = (float*)(ws + 18740224);    //    46,080 B

  const int packA = MT1 * KS1 * 32 * 16;    // 2,359,296 (same count for A2)
  pack_x_a1<<<(packA + 255) / 256, 256, 0, stream>>>(x, XP);
  pack_x_a2<<<(packA + 255) / 256, 256, 0, stream>>>(x, XTP);
  zero_k<<<(I_ * N_ + 255) / 256, 256, 0, stream>>>(bij, I_ * N_);

  for (int it = 0; it < 3; ++it) {
    softmax_k<<<(I_ + 63) / 64, 64, 0, stream>>>(bij, c);
    const int wcN = N_ * KS1 * 32 * 16;     // 1,474,560
    pack_wc<<<(wcN + 255) / 256, 256, 0, stream>>>(W, c, WCP);
    gemm_s<<<dim3(MT1, N_), 32 * KSPLIT, 0, stream>>>(XP, WCP, s);
    squash_k<<<(B_ * NO + 255) / 256, 256, 0, stream>>>(
        s, v, (it == 2) ? (float*)d_out : nullptr);
    if (it < 2) {
      const int vpN = N_ * KS2 * 32 * 16;   // 40,960
      pack_v<<<(vpN + 255) / 256, 256, 0, stream>>>(v, VP);
      gemm_p<<<(MT2 * N_) / 4, 128, 0, stream>>>(XTP, VP, P);
      uvj_k<<<(I_ * N_ + 255) / 256, 256, 0, stream>>>(W, P, bij);
    }
  }
}
